// MSA1_69114613728826
// MI455X (gfx1250) — compile-verified, matrix operand order checked by reading, not run on hardware
//
#include <hip/hip_runtime.h>
#include <hip/hip_bf16.h>
#include <stdint.h>

// ---------------- problem constants ----------------
constexpr int BATCH = 8;
constexpr int SEQ   = 4096;
constexpr int NHEAD = 16;
constexpr int HDIM  = 64;
constexpr int NHID  = 1024;              // == NHEAD*HDIM
constexpr int MROWS = BATCH * SEQ;       // 32768
constexpr int NQK   = 2 * NHID;          // fused q|k output width

// ---------------- vector types ----------------
typedef __attribute__((ext_vector_type(16))) _Float16 v16h;
typedef __attribute__((ext_vector_type(8)))  _Float16 v8h;
typedef __attribute__((ext_vector_type(4)))  _Float16 v4h;
typedef __attribute__((ext_vector_type(2)))  _Float16 v2h;
typedef __attribute__((ext_vector_type(8)))  float    v8f;

typedef union { v8f v; float f[8]; } v8f_u;

// ---------------- workspace layout (bytes) ----------------
constexpr size_t OFF_XH   = 0;                                    // 32768x1024 f16 (64 MB)
constexpr size_t OFF_WQKT = OFF_XH   + (size_t)MROWS * NHID * 2;  // 2048x1024 f16 ([WQ^T;WK^T])
constexpr size_t OFF_WOT  = OFF_WQKT + (size_t)NQK   * NHID * 2;  // 1024x1024 f16
constexpr size_t OFF_BQK  = OFF_WOT  + (size_t)NHID  * NHID * 2;  // 2048 f32
constexpr size_t OFF_QKH  = OFF_BQK  + (size_t)NQK * 4;           // 32768x2048 f16 (128 MB)
constexpr size_t OFF_ATTH = OFF_QKH  + (size_t)MROWS * NQK * 2;   // 32768x1024 f16
constexpr size_t OFF_AK   = OFF_ATTH + (size_t)MROWS * NHID * 2;  // 8x1024 f32
constexpr size_t OFF_AV   = OFF_AK   + (size_t)BATCH * NHID * 4;

// =====================================================================
// f32 -> f16 conversion, 4 elements / thread
// =====================================================================
__global__ void cvt_f32_to_f16_x4(const float* __restrict__ in,
                                  _Float16* __restrict__ out, int n4) {
  int i = blockIdx.x * blockDim.x + threadIdx.x;
  if (i >= n4) return;
  float4 v = ((const float4*)in)[i];
  v4h h;
  h[0] = (_Float16)v.x; h[1] = (_Float16)v.y;
  h[2] = (_Float16)v.z; h[3] = (_Float16)v.w;
  ((v4h*)out)[i] = h;
}

// =====================================================================
// f32 [K][N] -> f16 [N][K] transpose (32x32 LDS tiles)
// =====================================================================
__global__ void transpose_f32_to_f16(const float* __restrict__ in,
                                     _Float16* __restrict__ out,
                                     int K, int N) {
  __shared__ float tile[32][33];
  int tx = threadIdx.x, ty = threadIdx.y;
  int k0 = blockIdx.y * 32, n0 = blockIdx.x * 32;
#pragma unroll
  for (int i = 0; i < 32; i += 8)
    tile[ty + i][tx] = in[(size_t)(k0 + ty + i) * N + n0 + tx];
  __syncthreads();
#pragma unroll
  for (int i = 0; i < 32; i += 8)
    out[(size_t)(n0 + ty + i) * K + k0 + tx] = (_Float16)tile[tx][ty + i];
}

// =====================================================================
// concat two f32 vectors of length n -> out[2n]
// =====================================================================
__global__ void concat_bias(const float* __restrict__ a, const float* __restrict__ b,
                            float* __restrict__ out, int n) {
  int i = blockIdx.x * blockDim.x + threadIdx.x;
  out[i] = (i < n) ? a[i] : b[i - n];
}

// =====================================================================
// tiny projection for the global token: out[b][n] = ax[b]·W[:,n] + bias[n]
// =====================================================================
__global__ void proj_global_tok(const float* __restrict__ ax,
                                const float* __restrict__ W,
                                const float* __restrict__ bias,
                                float* __restrict__ out) {
  int n = blockIdx.x * blockDim.x + threadIdx.x;
  int b = blockIdx.y;
  const float* axr = ax + (size_t)b * NHID;
  float acc = bias[n];
#pragma unroll 4
  for (int k = 0; k < NHID; ++k)
    acc += axr[k] * W[(size_t)k * NHID + n];
  out[(size_t)b * NHID + n] = acc;
}

// =====================================================================
// WMMA GEMM:  C[M][N] = A[M][K](f16) * Bt[N][K](f16)^T + bias[N]
// 256 threads = 8 waves; block tile 64(M) x 256(N); K-step 32.
// Double-buffered LDS fed by GLOBAL_LOAD_ASYNC_TO_LDS_B128 (ASYNCcnt),
// no VGPR staging, 1 barrier / K-step.
// Each wave: 32x64 tile -> 2x4 v_wmma_f32_16x16x32_f16 per K-step.
// =====================================================================
constexpr int BK  = 32;
constexpr int LDT = BK + 8;    // padded LDS row (halves), preserves 16B alignment

template <bool OUT_F16>
__global__ __launch_bounds__(256, 1)
void gemm_wmma_f16(const _Float16* __restrict__ A,
                   const _Float16* __restrict__ Bt,
                   const float* __restrict__ bias,
                   void* __restrict__ Cout,
                   int M, int N, int K) {
  __shared__ _Float16 As[2][64 * LDT];
  __shared__ _Float16 Bs[2][256 * LDT];

  const int tid  = threadIdx.x;
  const int wid  = tid >> 5;
  const int lane = tid & 31;
  const int lh   = lane & 15;
  const int hi   = lane >> 4;          // 0: lanes 0-15, 1: lanes 16-31

  const int wm = wid >> 2;             // 0..1  (32 M-rows each)
  const int wn = wid & 3;              // 0..3  (64 N-cols each)

  const int m0 = blockIdx.y * 64;
  const int n0 = blockIdx.x * 256;

  const int ldrow = tid >> 2;          // 0..63
  const int ldchk = tid & 3;           // 0..3  (8-half chunks)

  const _Float16* Agp = A  + (size_t)(m0 + ldrow) * K + ldchk * 8;
  const _Float16* Bgp = Bt + (size_t)(n0 + ldrow) * K + ldchk * 8;
  const size_t Brow = (size_t)64 * K;  // 64-row stride between B passes

  // per-thread LDS destination addresses (low 32 bits of shared pointer =
  // wave-relative LDS byte address; HW adds LDS_BASE)
  uint32_t ldsA[2], ldsB[2];
#pragma unroll
  for (int i = 0; i < 2; ++i) {
    ldsA[i] = (uint32_t)(uintptr_t)&As[i][ldrow * LDT + ldchk * 8];
    ldsB[i] = (uint32_t)(uintptr_t)&Bs[i][ldrow * LDT + ldchk * 8];
  }

  // async global -> LDS staging of one K-tile (1 A chunk + 4 B chunks)
  auto issue_tile = [&](int buf, int k0) {
    asm volatile("global_load_async_to_lds_b128 %0, %1, off"
                 :: "v"(ldsA[buf]),
                    "v"((uint64_t)(uintptr_t)(Agp + k0))
                 : "memory");
#pragma unroll
    for (int p = 0; p < 4; ++p) {
      asm volatile("global_load_async_to_lds_b128 %0, %1, off"
                   :: "v"((uint32_t)(ldsB[buf] + p * (64 * LDT * 2))),
                      "v"((uint64_t)(uintptr_t)(Bgp + p * Brow + k0))
                   : "memory");
    }
  };

  v8f_u acc[2][4];
#pragma unroll
  for (int i = 0; i < 2; ++i)
#pragma unroll
    for (int j = 0; j < 4; ++j) acc[i][j].v = (v8f)(0.0f);

  const int nkt = K / BK;

  issue_tile(0, 0);                       // prologue: tile 0 in flight

  for (int kt = 0; kt < nkt; ++kt) {
    const int cur = kt & 1, nxt = cur ^ 1;

    // drain my async writes to lds[cur]; barrier makes all waves' writes
    // visible and guarantees every reader of lds[nxt] (iter kt-1) is done.
    asm volatile("s_wait_asynccnt 0x0" ::: "memory");
    __syncthreads();

    if (kt + 1 < nkt) {
      issue_tile(nxt, (kt + 1) * BK);     // in flight during this K-step
      if (kt + 2 < nkt)
        __builtin_prefetch(Agp + (kt + 2) * BK, 0, 1);
    }

    // ---- gather all fragments, then a back-to-back WMMA burst ----
    v16h aF[2], bF[4];
#pragma unroll
    for (int mf = 0; mf < 2; ++mf) {
      const int m = wm * 32 + mf * 16 + lh;
      const v8h lo = *(const v8h*)&As[cur][m * LDT + hi * 8];       // K 0..7 / 8..15
      const v8h hh = *(const v8h*)&As[cur][m * LDT + 16 + hi * 8];  // K 16..23 / 24..31
      aF[mf] = __builtin_shufflevector(lo, hh,
                 0,1,2,3,4,5,6,7, 8,9,10,11,12,13,14,15);
    }
#pragma unroll
    for (int nf = 0; nf < 4; ++nf) {
      const int n = wn * 64 + nf * 16 + lh;
      const v8h lo = *(const v8h*)&Bs[cur][n * LDT + hi * 8];
      const v8h hh = *(const v8h*)&Bs[cur][n * LDT + 16 + hi * 8];
      bF[nf] = __builtin_shufflevector(lo, hh,
                 0,1,2,3,4,5,6,7, 8,9,10,11,12,13,14,15);
    }
#pragma unroll
    for (int nf = 0; nf < 4; ++nf)
#pragma unroll
      for (int mf = 0; mf < 2; ++mf)
        acc[mf][nf].v = __builtin_amdgcn_wmma_f32_16x16x32_f16(
            false, aF[mf], false, bF[nf], (short)0, acc[mf][nf].v, false, false);
  }

  // ---- epilogue: C layout = VGPR r -> (row r + 8*hi, col lane&15) ----
#pragma unroll
  for (int nf = 0; nf < 4; ++nf) {
    const int col  = n0 + wn * 64 + nf * 16 + lh;
    const float bc = bias[col];
#pragma unroll
    for (int mf = 0; mf < 2; ++mf) {
      const int rbase = m0 + wm * 32 + mf * 16 + hi * 8;
#pragma unroll
      for (int r = 0; r < 8; ++r) {
        const float val = acc[mf][nf].f[r] + bc;
        const size_t idx = (size_t)(rbase + r) * N + col;
        if (OUT_F16) ((_Float16*)Cout)[idx] = (_Float16)val;
        else         ((float*)Cout)[idx]    = val;
      }
    }
  }
}

// =====================================================================
// attention epilogue: one wave per (b, l, h); lane owns 2 dims.
// qk buffer holds q (cols 0..1023) and k (cols 1024..2047) per token.
// slots: [ak, k_{l-1}, k_l, k_{l+1}]; V = [av, k_{l-1}, k_l, k_{l+1}]
// boundary slots are zero vectors -> score 0 (matches zero-padding).
// =====================================================================
__global__ __launch_bounds__(256)
void attn_window(const _Float16* __restrict__ qkh,
                 const float* __restrict__ akf,
                 const float* __restrict__ avf,
                 _Float16* __restrict__ atth) {
  const int gw   = blockIdx.x * (blockDim.x >> 5) + (threadIdx.x >> 5);
  const int lane = threadIdx.x & 31;
  const int h = gw & (NHEAD - 1);
  const int t = gw >> 4;              // b*SEQ + l
  const int l = t & (SEQ - 1);

  const size_t rowq = (size_t)t * NQK + h * HDIM + lane * 2;
  const size_t rowk = rowq + NHID;
  const size_t grow = (size_t)(t >> 12) * NHID + h * HDIM + lane * 2; // b*NHID + ...

  const v2h qv  = *(const v2h*)(qkh + rowq);
  const v2h kcv = *(const v2h*)(qkh + rowk);
  v2h kmv = {}; if (l > 0)       kmv = *(const v2h*)(qkh + rowk - NQK);
  v2h kpv = {}; if (l < SEQ - 1) kpv = *(const v2h*)(qkh + rowk + NQK);
  const float2 ak2 = *(const float2*)(akf + grow);
  const float2 av2 = *(const float2*)(avf + grow);

  const float q0 = (float)qv[0], q1 = (float)qv[1];
  float s0 = q0 * ak2.x          + q1 * ak2.y;
  float s1 = q0 * (float)kmv[0]  + q1 * (float)kmv[1];
  float s2 = q0 * (float)kcv[0]  + q1 * (float)kcv[1];
  float s3 = q0 * (float)kpv[0]  + q1 * (float)kpv[1];

#pragma unroll
  for (int off = 16; off > 0; off >>= 1) {
    s0 += __shfl_xor(s0, off, 32);
    s1 += __shfl_xor(s1, off, 32);
    s2 += __shfl_xor(s2, off, 32);
    s3 += __shfl_xor(s3, off, 32);
  }
  const float sc = 0.125f;  // 1/sqrt(64)
  s0 *= sc; s1 *= sc; s2 *= sc; s3 *= sc;

  const float mx = fmaxf(fmaxf(s0, s1), fmaxf(s2, s3));
  const float e0 = __expf(s0 - mx), e1 = __expf(s1 - mx);
  const float e2 = __expf(s2 - mx), e3 = __expf(s3 - mx);
  const float inv = 1.0f / (e0 + e1 + e2 + e3);
  const float a0 = e0 * inv, a1 = e1 * inv, a2 = e2 * inv, a3 = e3 * inv;

  const float o0 = a0 * av2.x + a1 * (float)kmv[0] + a2 * (float)kcv[0] + a3 * (float)kpv[0];
  const float o1 = a0 * av2.y + a1 * (float)kmv[1] + a2 * (float)kcv[1] + a3 * (float)kpv[1];

  v2h ov; ov[0] = (_Float16)o0; ov[1] = (_Float16)o1;
  *(v2h*)(atth + (size_t)t * NHID + h * HDIM + lane * 2) = ov;
}

// =====================================================================
// launch
// =====================================================================
extern "C" void kernel_launch(void* const* d_in, const int* in_sizes, int n_in,
                              void* d_out, int out_size, void* d_ws, size_t ws_size,
                              hipStream_t stream) {
  const float* x    = (const float*)d_in[0];
  const float* ax   = (const float*)d_in[1];
  const float* WQ_w = (const float*)d_in[2];
  const float* WQ_b = (const float*)d_in[3];
  const float* WK_w = (const float*)d_in[4];
  const float* WK_b = (const float*)d_in[5];
  const float* WV_w = (const float*)d_in[6];
  const float* WV_b = (const float*)d_in[7];
  const float* WO_w = (const float*)d_in[8];
  const float* WO_b = (const float*)d_in[9];

  char* ws = (char*)d_ws;
  _Float16* xh   = (_Float16*)(ws + OFF_XH);
  _Float16* wqkt = (_Float16*)(ws + OFF_WQKT);
  _Float16* wot  = (_Float16*)(ws + OFF_WOT);
  float*    bqk  = (float*)(ws + OFF_BQK);
  _Float16* qkh  = (_Float16*)(ws + OFF_QKH);
  _Float16* atth = (_Float16*)(ws + OFF_ATTH);
  float*    akf  = (float*)(ws + OFF_AK);
  float*    avf  = (float*)(ws + OFF_AV);

  // 1) convert x -> f16
  {
    int n4 = MROWS * NHID / 4;
    cvt_f32_to_f16_x4<<<(n4 + 255) / 256, 256, 0, stream>>>(x, xh, n4);
  }
  // 2) transpose+convert weights: [WQ^T ; WK^T] fused, and WO^T
  {
    dim3 g(NHID / 32, NHID / 32), b(32, 8);
    transpose_f32_to_f16<<<g, b, 0, stream>>>(WQ_w, wqkt, NHID, NHID);
    transpose_f32_to_f16<<<g, b, 0, stream>>>(WK_w, wqkt + (size_t)NHID * NHID, NHID, NHID);
    transpose_f32_to_f16<<<g, b, 0, stream>>>(WO_w, wot, NHID, NHID);
  }
  // 3) fused bias [WQ_b ; WK_b]
  concat_bias<<<NQK / 256, 256, 0, stream>>>(WQ_b, WK_b, bqk, NHID);
  // 4) global-token projections ak = ax@WK + bK, av = ax@WV + bV (f32)
  {
    dim3 g(NHID / 256, BATCH);
    proj_global_tok<<<g, 256, 0, stream>>>(ax, WK_w, WK_b, akf);
    proj_global_tok<<<g, 256, 0, stream>>>(ax, WV_w, WV_b, avf);
  }
  // 5) fused q|k projection via WMMA GEMM: [32768x1024] @ [1024x2048]
  {
    dim3 g(NQK / 256, MROWS / 64);
    gemm_wmma_f16<true><<<g, 256, 0, stream>>>(xh, wqkt, bqk, (void*)qkh,
                                               MROWS, NQK, NHID);
  }
  // 6) windowed attention epilogue -> att (f16)
  {
    int waves = BATCH * SEQ * NHEAD;           // 524288
    attn_window<<<waves / 8, 256, 0, stream>>>(qkh, akf, avf, atth);
  }
  // 7) output projection via WMMA GEMM (f32 out to d_out)
  {
    dim3 g(NHID / 256, MROWS / 64);
    gemm_wmma_f16<false><<<g, 256, 0, stream>>>(atth, wot, WO_b, d_out,
                                                MROWS, NHID, NHID);
  }
}
